// GraphGinEdgeNet_59966333387406
// MI455X (gfx1250) — compile-verified, matrix operand order checked by reading, not run on hardware
//
#include <hip/hip_runtime.h>
#include <hip/hip_bf16.h>

#define NNODES 50000
#define NEDGES 600000
#define CH 128
#define BNEPS 1e-5f
#define WPB 8   // waves per 256-thread block

typedef __attribute__((ext_vector_type(16))) __bf16 bf16x16;
typedef __attribute__((ext_vector_type(8)))  float  f32x8;

union BF16Pack { unsigned int u[8]; bf16x16 v; };

// packed f32x2 -> bf16x2 (lowers to v_cvt_pk_bf16_f32 on gfx1250)
__device__ __forceinline__ unsigned int pack2bf(float a, float b) {
  union { __hip_bfloat162 h; unsigned int u; } cv;
  cv.h = __float22bfloat162_rn(make_float2(a, b));
  return cv.u;
}
__device__ __forceinline__ unsigned short f2bf(float f) {
  union { __hip_bfloat16 h; unsigned short u; } cv;
  cv.h = __float2bfloat16(f);
  return cv.u;
}

__device__ __forceinline__ f32x8 wmma_bf16(bf16x16 a, bf16x16 b, f32x8 c) {
  return __builtin_amdgcn_wmma_f32_16x16x32_bf16(false, a, false, b, (short)0, c,
                                                 false, false);
}

// ---- A-tile chunk loaders (16x32 bf16 A-layout: lane<16 K=0..7,16..23; lane>=16 K=8..15,24..31)
// p points at row_start + kt*32 + kh8 (kh8 = (lane>>4)*8)
__device__ __forceinline__ bf16x16 load_a_f32(const float* p) {
  float4 a0 = *(const float4*)(p);
  float4 a1 = *(const float4*)(p + 4);
  float4 b0 = *(const float4*)(p + 16);
  float4 b1 = *(const float4*)(p + 20);
  BF16Pack r;
  r.u[0] = pack2bf(a0.x, a0.y); r.u[1] = pack2bf(a0.z, a0.w);
  r.u[2] = pack2bf(a1.x, a1.y); r.u[3] = pack2bf(a1.z, a1.w);
  r.u[4] = pack2bf(b0.x, b0.y); r.u[5] = pack2bf(b0.z, b0.w);
  r.u[6] = pack2bf(b1.x, b1.y); r.u[7] = pack2bf(b1.z, b1.w);
  return r.v;
}
__device__ __forceinline__ bf16x16 load_a_diff_f32(const float* pj, const float* pi) {
  float4 j0 = *(const float4*)(pj),      i0 = *(const float4*)(pi);
  float4 j1 = *(const float4*)(pj + 4),  i1 = *(const float4*)(pi + 4);
  float4 j2 = *(const float4*)(pj + 16), i2 = *(const float4*)(pi + 16);
  float4 j3 = *(const float4*)(pj + 20), i3 = *(const float4*)(pi + 20);
  BF16Pack r;
  r.u[0] = pack2bf(j0.x - i0.x, j0.y - i0.y); r.u[1] = pack2bf(j0.z - i0.z, j0.w - i0.w);
  r.u[2] = pack2bf(j1.x - i1.x, j1.y - i1.y); r.u[3] = pack2bf(j1.z - i1.z, j1.w - i1.w);
  r.u[4] = pack2bf(j2.x - i2.x, j2.y - i2.y); r.u[5] = pack2bf(j2.z - i2.z, j2.w - i2.w);
  r.u[6] = pack2bf(j3.x - i3.x, j3.y - i3.y); r.u[7] = pack2bf(j3.z - i3.z, j3.w - i3.w);
  return r.v;
}
// 16 bf16 from memory (weights already bf16, row-major [Cout,K]); p 16B-aligned
__device__ __forceinline__ bf16x16 load_b_bf16(const unsigned short* p) {
  uint4 lo = *(const uint4*)p;
  uint4 hi = *(const uint4*)(p + 16);
  BF16Pack r;
  r.u[0] = lo.x; r.u[1] = lo.y; r.u[2] = lo.z; r.u[3] = lo.w;
  r.u[4] = hi.x; r.u[5] = hi.y; r.u[6] = hi.z; r.u[7] = hi.w;
  return r.v;
}

// ---------------------------------------------------------------------------
// Weight prep: bf16 convert with optional BatchNorm folding into the Linear.
// W' = diag(s) W,  b' = s*(b - m) + beta,  s = g/sqrt(v+eps)
// ---------------------------------------------------------------------------
__global__ void prep_weights(const float* __restrict__ W, const float* __restrict__ B,
                             const float* __restrict__ bng, const float* __restrict__ bnb,
                             const float* __restrict__ bnm, const float* __restrict__ bnv,
                             unsigned short* __restrict__ Wo, float* __restrict__ Bo,
                             int Cout, int K, int doBN) {
  int total = Cout * K;
  for (int idx = blockIdx.x * blockDim.x + threadIdx.x; idx < total;
       idx += gridDim.x * blockDim.x) {
    int c = idx / K;
    float s = doBN ? bng[c] * rsqrtf(bnv[c] + BNEPS) : 1.0f;
    Wo[idx] = f2bf(W[idx] * s);
    if (idx < Cout) {
      float bb = doBN ? s * (B[idx] - bnm[idx]) + bnb[idx] : B[idx];
      Bo[idx] = bb;
    }
  }
}

__global__ void zero_f32(float* __restrict__ p, int n) {
  for (int i = blockIdx.x * blockDim.x + threadIdx.x; i < n; i += gridDim.x * blockDim.x)
    p[i] = 0.0f;
}

__global__ void count_kernel(const int* __restrict__ ei, float* __restrict__ cnt, int nE) {
  int e = blockIdx.x * blockDim.x + threadIdx.x;
  if (e < nE) atomicAdd(&cnt[ei[nE + e]], 1.0f);
}

// agg[dst] += g[src], 4 channels per thread
__global__ void scatter_rows(const float* __restrict__ G, const int* __restrict__ ei,
                             float* __restrict__ S, int nE) {
  int t = blockIdx.x * blockDim.x + threadIdx.x;
  if (t >= nE * 32) return;
  int e = t >> 5;
  int c = (t & 31) << 2;
  int s = ei[e], d = ei[nE + e];
  float4 v = *(const float4*)(G + s * CH + c);
  float* o = S + d * CH + c;
  atomicAdd(o + 0, v.x); atomicAdd(o + 1, v.y);
  atomicAdd(o + 2, v.z); atomicAdd(o + 3, v.w);
}

// S = (1+eps)*G + S   (GIN pre-MLP, in-place on agg buffer)
__global__ void gin_pre(const float* __restrict__ G, float* __restrict__ S,
                        const float* __restrict__ epsp, int n) {
  float e1 = 1.0f + *epsp;
  for (int i = blockIdx.x * blockDim.x + threadIdx.x; i < n; i += gridDim.x * blockDim.x)
    S[i] = e1 * G[i] + S[i];
}

// Y = maybe_relu(maybe_bn( S/max(cnt,1) + H ))
__global__ void combine_kernel(const float* __restrict__ S, const float* __restrict__ cnt,
                               const float* __restrict__ H, float* __restrict__ Y,
                               const float* __restrict__ bg, const float* __restrict__ bb,
                               const float* __restrict__ bm, const float* __restrict__ bv,
                               int doBN, int n) {
  for (int i = blockIdx.x * blockDim.x + threadIdx.x; i < n; i += gridDim.x * blockDim.x) {
    int node = i >> 7;
    int c = i & (CH - 1);
    float cn = cnt[node]; cn = cn > 1.0f ? cn : 1.0f;
    float y = S[i] / cn + H[i];
    if (doBN) {
      float s = bg[c] * rsqrtf(bv[c] + BNEPS);
      y = fmaxf((y - bm[c]) * s + bb[c], 0.0f);
    }
    Y[i] = y;
  }
}

// ---------------------------------------------------------------------------
// Node GEMM: Y[N,128] = maybe_relu(X[N,K] @ W^T + b), W bf16 [128,K], one wave
// per 16-row tile, A preloaded before any store (X may equal Y: no restrict).
// ---------------------------------------------------------------------------
template <int KT>  // K = KT*32
__global__ __launch_bounds__(256)
void gemm_node_kernel(const float* X, int ldx,
                      const unsigned short* __restrict__ W,
                      const float* __restrict__ bias,
                      float* Y, int nrows, int ntiles, int doRelu) {
  int wave = (blockIdx.x * blockDim.x + threadIdx.x) >> 5;
  if (wave >= ntiles) return;
  int lane = threadIdx.x & 31;
  int mlane = lane & 15;
  int kh8 = (lane >> 4) << 3;
  int row = wave * 16 + mlane;
  int arow = row < nrows ? row : nrows - 1;
  const int K = KT * 32;

  bf16x16 a[KT];
  const float* rp = X + (long)arow * ldx + kh8;
#pragma unroll
  for (int kt = 0; kt < KT; ++kt) a[kt] = load_a_f32(rp + kt * 32);

  int rbase = wave * 16 + ((lane >> 4) << 3);
#pragma unroll
  for (int nt = 0; nt < 8; ++nt) {
    int col = nt * 16 + mlane;
    float bv = bias[col];
    f32x8 c;
#pragma unroll
    for (int r = 0; r < 8; ++r) c[r] = bv;
#pragma unroll
    for (int kt = 0; kt < KT; ++kt)
      c = wmma_bf16(a[kt], load_b_bf16(W + col * K + kt * 32 + kh8), c);
#pragma unroll
    for (int r = 0; r < 8; ++r) {
      float v = c[r];
      if (doRelu) v = fmaxf(v, 0.0f);
      int rr = rbase + r;
      if (rr < nrows) Y[rr * CH + col] = v;
    }
  }
}

// ---------------------------------------------------------------------------
// EdgeConv: per 16-edge tile: u = [h[dst], h[src]-h[dst]] (256) ->
// relu(bnfold(fc1)) -> LDS transpose (bf16) -> fc2 -> atomic scatter to S[dst].
// ---------------------------------------------------------------------------
__global__ __launch_bounds__(256)
void edge_conv_kernel(const float* __restrict__ H, const int* __restrict__ ei,
                      const unsigned short* __restrict__ W1,
                      const float* __restrict__ B1,
                      const unsigned short* __restrict__ W2,
                      const float* __restrict__ B2,
                      float* __restrict__ S, int nE, int ntiles) {
  __shared__ unsigned short lds[WPB * 16 * CH];
  int wib = threadIdx.x >> 5;
  int lane = threadIdx.x & 31;
  int tileRaw = blockIdx.x * WPB + wib;
  int active = tileRaw < ntiles;
  int tile = active ? tileRaw : ntiles - 1;
  unsigned short* tl = lds + wib * (16 * CH);
  int mlane = lane & 15;
  int kh8 = (lane >> 4) << 3;

  // warm L1/L2 with fc2 weights while fc1 runs
  __builtin_prefetch(W2 + (lane << 9), 0, 1);

  int e = tile * 16 + mlane;
  int di = ei[nE + e];  // dst (target i)
  int si = ei[e];       // src (source j)
  const float* hi = H + di * CH;
  const float* hj = H + si * CH;

  bf16x16 a[8];
#pragma unroll
  for (int kt = 0; kt < 4; ++kt) a[kt] = load_a_f32(hi + kt * 32 + kh8);
#pragma unroll
  for (int kt = 0; kt < 4; ++kt)
    a[4 + kt] = load_a_diff_f32(hj + kt * 32 + kh8, hi + kt * 32 + kh8);

  int rbase = (lane >> 4) << 3;
  // fc1 (K=256) with folded BN, ReLU, store bf16 to LDS in row-major [16 x 128]
#pragma unroll
  for (int nt = 0; nt < 8; ++nt) {
    int col = nt * 16 + mlane;
    float bv = B1[col];
    f32x8 c;
#pragma unroll
    for (int r = 0; r < 8; ++r) c[r] = bv;
#pragma unroll
    for (int kt = 0; kt < 8; ++kt)
      c = wmma_bf16(a[kt], load_b_bf16(W1 + col * 256 + kt * 32 + kh8), c);
#pragma unroll
    for (int r = 0; r < 8; ++r)
      tl[(rbase + r) * CH + col] = f2bf(fmaxf(c[r], 0.0f));
  }

  __syncthreads();  // order LDS writes (fc1 D-layout) before reads (fc2 A-layout)

  // reload as A-layout for fc2 (K=128)
  bf16x16 t[4];
#pragma unroll
  for (int kt = 0; kt < 4; ++kt)
    t[kt] = load_b_bf16(tl + mlane * CH + kt * 32 + kh8);

  int dstv[8];
  int ebase = tile * 16 + rbase;
#pragma unroll
  for (int r = 0; r < 8; ++r) dstv[r] = ei[nE + ebase + r];

#pragma unroll
  for (int nt = 0; nt < 8; ++nt) {
    int col = nt * 16 + mlane;
    float bv = B2[col];
    f32x8 c;
#pragma unroll
    for (int r = 0; r < 8; ++r) c[r] = bv;
#pragma unroll
    for (int kt = 0; kt < 4; ++kt)
      c = wmma_bf16(t[kt], load_b_bf16(W2 + col * CH + kt * 32 + kh8), c);
    if (active) {
#pragma unroll
      for (int r = 0; r < 8; ++r) atomicAdd(&S[dstv[r] * CH + col], c[r]);
    }
  }
}

// final Linear 128 -> 2
__global__ void fc_out_kernel(const float* __restrict__ G, const float* __restrict__ W,
                              const float* __restrict__ B, float* __restrict__ out, int n) {
  int i = blockIdx.x * blockDim.x + threadIdx.x;
  if (i >= n) return;
  const float* x = G + i * CH;
  float a0 = B[0], a1 = B[1];
#pragma unroll
  for (int c = 0; c < CH; c += 4) {
    float4 xv = *(const float4*)(x + c);
    float4 w0 = *(const float4*)(W + c);
    float4 w1 = *(const float4*)(W + CH + c);
    a0 += xv.x * w0.x + xv.y * w0.y + xv.z * w0.z + xv.w * w0.w;
    a1 += xv.x * w1.x + xv.y * w1.y + xv.z * w1.z + xv.w * w1.w;
  }
  out[i * 2 + 0] = a0;
  out[i * 2 + 1] = a1;
}

// ---------------------------------------------------------------------------
extern "C" void kernel_launch(void* const* d_in, const int* in_sizes, int n_in,
                              void* d_out, int out_size, void* d_ws, size_t ws_size,
                              hipStream_t stream) {
  const float* x      = (const float*)d_in[0];   // [N,2,128] == [N,256]
  const int*   ei     = (const int*)d_in[1];     // [2,E]
  const float* av_w   = (const float*)d_in[2];
  const float* av_b   = (const float*)d_in[3];
  const float* gineps = (const float*)d_in[4];
  const float* g1w = (const float*)d_in[5],  *g1b = (const float*)d_in[6];
  const float* gbg = (const float*)d_in[7],  *gbb = (const float*)d_in[8];
  const float* gbm = (const float*)d_in[9],  *gbv = (const float*)d_in[10];
  const float* g2w = (const float*)d_in[11], *g2b = (const float*)d_in[12];
  const float* b1g = (const float*)d_in[13], *b1b = (const float*)d_in[14];
  const float* b1m = (const float*)d_in[15], *b1v = (const float*)d_in[16];
  const float* e1w[3] = {(const float*)d_in[17], (const float*)d_in[29], (const float*)d_in[41]};
  const float* e1bv[3]= {(const float*)d_in[18], (const float*)d_in[30], (const float*)d_in[42]};
  const float* ebg[3] = {(const float*)d_in[19], (const float*)d_in[31], (const float*)d_in[43]};
  const float* ebb[3] = {(const float*)d_in[20], (const float*)d_in[32], (const float*)d_in[44]};
  const float* ebm[3] = {(const float*)d_in[21], (const float*)d_in[33], (const float*)d_in[45]};
  const float* ebv[3] = {(const float*)d_in[22], (const float*)d_in[34], (const float*)d_in[46]};
  const float* e2w[3] = {(const float*)d_in[23], (const float*)d_in[35], (const float*)d_in[47]};
  const float* e2bv[3]= {(const float*)d_in[24], (const float*)d_in[36], (const float*)d_in[48]};
  const float* pbg[2] = {(const float*)d_in[25], (const float*)d_in[37]};
  const float* pbb[2] = {(const float*)d_in[26], (const float*)d_in[38]};
  const float* pbm[2] = {(const float*)d_in[27], (const float*)d_in[39]};
  const float* pbv[2] = {(const float*)d_in[28], (const float*)d_in[40]};
  const float* fcw = (const float*)d_in[49], *fcb = (const float*)d_in[50];
  float* out = (float*)d_out;

  // ---- workspace carve-up
  char* base = (char*)d_ws;
  size_t off = 0;
  auto carve = [&](size_t bytes) { void* p = base + off; off += (bytes + 255) & ~(size_t)255; return p; };
  const size_t NB = (size_t)NNODES * CH * sizeof(float);
  float* bufA = (float*)carve(NB);
  float* bufB = (float*)carve(NB);
  float* bufS = (float*)carve(NB);
  float* cnt  = (float*)carve((size_t)NNODES * sizeof(float));
  unsigned short* wav = (unsigned short*)carve(128 * 256 * 2);
  unsigned short* wg1 = (unsigned short*)carve(128 * 128 * 2);
  unsigned short* wg2 = (unsigned short*)carve(128 * 128 * 2);
  unsigned short* we1[3], *we2[3];
  for (int l = 0; l < 3; ++l) we1[l] = (unsigned short*)carve(128 * 256 * 2);
  for (int l = 0; l < 3; ++l) we2[l] = (unsigned short*)carve(128 * 128 * 2);
  float* bav = (float*)carve(128 * 4);
  float* bg1 = (float*)carve(128 * 4);
  float* bg2 = (float*)carve(128 * 4);
  float* be1[3], *be2[3];
  for (int l = 0; l < 3; ++l) be1[l] = (float*)carve(128 * 4);
  for (int l = 0; l < 3; ++l) be2[l] = (float*)carve(128 * 4);
  (void)ws_size; (void)n_in; (void)in_sizes; (void)out_size;

  const int nElem = NNODES * CH;
  const int nodeTiles = (NNODES + 15) / 16;
  const int edgeTiles = NEDGES / 16;
  dim3 blk(256);

  // ---- 1. weight prep (BN folding + bf16)
  {
    int gw = (128 * 256 + 255) / 256;
    prep_weights<<<gw, blk, 0, stream>>>(av_w, av_b, nullptr, nullptr, nullptr, nullptr,
                                         wav, bav, 128, 256, 0);
    int gs = (128 * 128 + 255) / 256;
    prep_weights<<<gs, blk, 0, stream>>>(g1w, g1b, gbg, gbb, gbm, gbv, wg1, bg1, 128, 128, 1);
    prep_weights<<<gs, blk, 0, stream>>>(g2w, g2b, b1g, b1b, b1m, b1v, wg2, bg2, 128, 128, 1);
    for (int l = 0; l < 3; ++l) {
      prep_weights<<<gw, blk, 0, stream>>>(e1w[l], e1bv[l], ebg[l], ebb[l], ebm[l], ebv[l],
                                           we1[l], be1[l], 128, 256, 1);
      prep_weights<<<gs, blk, 0, stream>>>(e2w[l], e2bv[l], nullptr, nullptr, nullptr, nullptr,
                                           we2[l], be2[l], 128, 128, 0);
    }
  }

  // ---- 2. degree counts (shared by all EdgeConv layers)
  zero_f32<<<(NNODES + 255) / 256, blk, 0, stream>>>(cnt, NNODES);
  count_kernel<<<(NEDGES + 255) / 256, blk, 0, stream>>>(ei, cnt, NEDGES);

  // ---- 3. av_fusion: g = x[N,256] @ Wav^T + b   -> bufA
  gemm_node_kernel<8><<<(nodeTiles + WPB - 1) / WPB, blk, 0, stream>>>(
      x, 256, wav, bav, bufA, NNODES, nodeTiles, 0);

  // ---- 4. GIN: agg = segsum(g[src] -> dst); h = LPP((1+eps)g + agg); bn1+relu folded
  zero_f32<<<4096, blk, 0, stream>>>(bufS, nElem);
  scatter_rows<<<(NEDGES * 32 + 255) / 256, blk, 0, stream>>>(bufA, ei, bufS, NEDGES);
  gin_pre<<<4096, blk, 0, stream>>>(bufA, bufS, gineps, nElem);
  gemm_node_kernel<4><<<(nodeTiles + WPB - 1) / WPB, blk, 0, stream>>>(
      bufS, CH, wg1, bg1, bufS, NNODES, nodeTiles, 1);          // fc1+bn+relu, in-place
  gemm_node_kernel<4><<<(nodeTiles + WPB - 1) / WPB, blk, 0, stream>>>(
      bufS, CH, wg2, bg2, bufB, NNODES, nodeTiles, 1);          // fc2+bn1+relu -> g1 in bufB

  // ---- 5. EdgeConv layers: h alternates bufB -> bufA -> bufB -> bufA
  float* hcur = bufB;
  float* hnxt = bufA;
  for (int l = 0; l < 3; ++l) {
    zero_f32<<<4096, blk, 0, stream>>>(bufS, nElem);
    edge_conv_kernel<<<(edgeTiles + WPB - 1) / WPB, blk, 0, stream>>>(
        hcur, ei, we1[l], be1[l], we2[l], be2[l], bufS, NEDGES, edgeTiles);
    if (l < 2) {
      combine_kernel<<<4096, blk, 0, stream>>>(bufS, cnt, hcur, hnxt,
                                               pbg[l], pbb[l], pbm[l], pbv[l], 1, nElem);
    } else {
      combine_kernel<<<4096, blk, 0, stream>>>(bufS, cnt, hcur, hnxt,
                                               nullptr, nullptr, nullptr, nullptr, 0, nElem);
    }
    float* t = hcur; hcur = hnxt; hnxt = t;
  }

  // ---- 6. final fc: out[N,2]
  fc_out_kernel<<<(NNODES + 255) / 256, blk, 0, stream>>>(hcur, fcw, fcb, out, NNODES);
}
